// Head_15779709845912
// MI455X (gfx1250) — matrix operand order repaired and checked by reading, not compiled, not run on hardware
//
#include <hip/hip_runtime.h>

#define DEVINL __device__ __forceinline__

typedef __attribute__((ext_vector_type(16))) __bf16 v16bf;
typedef __attribute__((ext_vector_type(8)))  __bf16 v8bf;
typedef __attribute__((ext_vector_type(8)))  float  v8f;

namespace {
constexpr int kB = 4, kT = 4096, kC = 1024, kH = 64;
constexpr int kRows = kB * kT;               // 16384
constexpr float kScale = 0.03125f;           // 1024^-0.5 (reference scales by C^-0.5)
}

DEVINL v8f wmma_bf16(v16bf a, v16bf b, v8f c) {
  // (neg_a, A, neg_b, B, c_mod, C, reuse_a, reuse_b)
  return __builtin_amdgcn_wmma_f32_16x16x32_bf16(false, a, false, b, (short)0, c,
                                                 false, false);
}

union V16U { v16bf v; v8bf h[2]; };

// ---------------------------------------------------------------------------
// Pass 0: transpose + convert the three 1024x64 weight matrices to bf16.
// wT[mat][h][c] = W[mat][c][h]   (so projection B-fragments are contiguous)
// ---------------------------------------------------------------------------
__global__ void wtrans_kernel(const float* __restrict__ Wq,
                              const float* __restrict__ Wk,
                              const float* __restrict__ Wv,
                              __bf16* __restrict__ wT) {
  int idx = blockIdx.x * blockDim.x + threadIdx.x;   // 0 .. 3*65536-1
  int mat = idx >> 16;
  int e   = idx & 65535;
  int c   = e >> 6;
  int h   = e & 63;
  const float* W = (mat == 0) ? Wq : (mat == 1) ? Wk : Wv;
  wT[mat * (kC * kH) + h * kC + c] = (__bf16)W[c * kH + h];
}

// ---------------------------------------------------------------------------
// Pass 1: QKV projection. [16384 x 1024] x [1024 x 64] with bf16 WMMA.
// One wave per 16-row tile; blockIdx.y selects Q / K / V.  x is L2-resident
// (64 MB vs 192 MB L2) so the 3x re-read is cheap and buys 3072 waves.
// Q,K stored row-major bf16 [row][h]; V stored transposed bf16 [h][row].
// ---------------------------------------------------------------------------
__global__ void __launch_bounds__(32) proj_kernel(
    const float* __restrict__ x, const __bf16* __restrict__ wT,
    __bf16* __restrict__ qbf, __bf16* __restrict__ kbf,
    __bf16* __restrict__ vT) {
  const int tile = blockIdx.x;
  const int mat  = blockIdx.y;               // 0=Q, 1=K, 2=V
  const int lane = threadIdx.x;
  const int m    = lane & 15;
  const int half = lane >> 4;
  const int row0 = tile * 16;

  const __bf16* wTm = wT + mat * (kC * kH);
  const float*  xr  = x + (size_t)(row0 + m) * kC;
  v8f acc[4] = {};

  for (int kb = 0; kb < kC; kb += 32) {
    // A fragment (16x32 bf16): lane<16 holds K {0..15}, lane>=16 {16..31}
    const int a0 = kb + half * 16;
    float4 f0 = *(const float4*)(xr + a0);
    float4 f1 = *(const float4*)(xr + a0 + 4);
    float4 f2 = *(const float4*)(xr + a0 + 8);
    float4 f3 = *(const float4*)(xr + a0 + 12);
    v16bf a;
    a[0]=(__bf16)f0.x;  a[1]=(__bf16)f0.y;  a[2]=(__bf16)f0.z;  a[3]=(__bf16)f0.w;
    a[4]=(__bf16)f1.x;  a[5]=(__bf16)f1.y;  a[6]=(__bf16)f1.z;  a[7]=(__bf16)f1.w;
    a[8]=(__bf16)f2.x;  a[9]=(__bf16)f2.y;  a[10]=(__bf16)f2.z; a[11]=(__bf16)f2.w;
    a[12]=(__bf16)f3.x; a[13]=(__bf16)f3.y; a[14]=(__bf16)f3.z; a[15]=(__bf16)f3.w;
#pragma unroll
    for (int c = 0; c < 4; ++c) {
      const int h = c * 16 + m;   // B fragment: lane = column h, contiguous K
      v16bf b = *(const v16bf*)(wTm + (size_t)h * kC + kb + half * 16);
      acc[c] = wmma_bf16(a, b, acc[c]);
    }
  }

  if (mat < 2) {
    __bf16* dst = (mat == 0) ? qbf : kbf;
#pragma unroll
    for (int c = 0; c < 4; ++c)
#pragma unroll
      for (int r = 0; r < 8; ++r)
        dst[(size_t)(row0 + r + half * 8) * kH + c * 16 + m] = (__bf16)acc[c][r];
  } else {
#pragma unroll
    for (int c = 0; c < 4; ++c)
#pragma unroll
      for (int r = 0; r < 8; ++r)
        vT[(size_t)(c * 16 + m) * kRows + row0 + r + half * 8] = (__bf16)acc[c][r];
  }
}

// ---------------------------------------------------------------------------
// Async DMA of one 32-key step: K tile (32x64 bf16) and V tile (64x32 bf16).
// 128 threads x 4 GLOBAL_LOAD_ASYNC_TO_LDS_B128 each (tracked by ASYNCcnt).
// ---------------------------------------------------------------------------
DEVINL void issue_tile_dma(const __bf16* kbf, const __bf16* vT,
                           __bf16* ldsk, __bf16* ldsv, int b, int jb, int tid) {
  const char* gk = (const char*)(kbf + (size_t)(b * kT + jb) * kH) + tid * 32;
  unsigned    lk = (unsigned)(uintptr_t)((char*)ldsk + tid * 32);
  asm volatile("global_load_async_to_lds_b128 %0, %1, off"
               :: "v"(lk), "v"(gk) : "memory");
  asm volatile("global_load_async_to_lds_b128 %0, %1, off offset:16"
               :: "v"(lk), "v"(gk) : "memory");

  const int vr = tid >> 1, vh = tid & 1;               // h-row, 32B half-row
  const char* gv = (const char*)(vT + (size_t)vr * kRows + b * kT + jb) + vh * 32;
  unsigned    lv = (unsigned)(uintptr_t)((char*)ldsv + vr * 64 + vh * 32);
  asm volatile("global_load_async_to_lds_b128 %0, %1, off"
               :: "v"(lv), "v"(gv) : "memory");
  asm volatile("global_load_async_to_lds_b128 %0, %1, off offset:16"
               :: "v"(lv), "v"(gv) : "memory");
}

// ---------------------------------------------------------------------------
// Pass 2: causal flash attention.
// 4 waves / block, each wave owns a 16-query tile of a 64-query block.
// K/V tiles are double-buffered in LDS: tile t+1 streams in via async DMA
// while tile t is consumed by bf16 WMMA + fp32 online softmax.  Async loads
// complete in order, so with 4 loads/thread/tile, ASYNCcnt<=4 <=> tile t
// resident.  P is re-shaped C/D-layout -> A-layout via a wave-private bounce.
// ---------------------------------------------------------------------------
__global__ void __launch_bounds__(128) attn_kernel(
    const __bf16* __restrict__ qbf, const __bf16* __restrict__ kbf,
    const __bf16* __restrict__ vT, float* __restrict__ out) {
  __shared__ __bf16 ldsk[2][32 * 64];      // [buf][key][h]      2 x 4 KB
  __shared__ __bf16 ldsv[2][64 * 32];      // [buf][h][key]      2 x 4 KB
  __shared__ __bf16 pshm[4][16 * 32];      // per-wave P bounce      4 KB

  const int blk   = blockIdx.x;            // 256 blocks
  const int b     = blk >> 6;              // 64 query blocks per batch
  const int iblk  = (blk & 63) * 64;       // 64-query block base within batch
  const int tid   = threadIdx.x;
  const int wave  = tid >> 5;
  const int lane  = tid & 31;
  const int m     = lane & 15;
  const int half  = (lane >> 4) & 1;
  const int ibase = iblk + wave * 16;      // this wave's query tile
  const int row0  = b * kT + ibase;

  // Q A-fragments for the two K=32 chunks of H=64 (held for the whole tile)
  V16U qa[2];
  const __bf16* qrow = qbf + (size_t)(row0 + m) * kH;
#pragma unroll
  for (int kk = 0; kk < 2; ++kk) {
    const int base = kk * 32 + half * 16;
    qa[kk].h[0] = *(const v8bf*)(qrow + base);
    qa[kk].h[1] = *(const v8bf*)(qrow + base + 8);
  }

  v8f acc[4] = {};
  float mrow[8], lrow[8];
#pragma unroll
  for (int r = 0; r < 8; ++r) { mrow[r] = -__builtin_inff(); lrow[r] = 0.0f; }

  const int njb = (iblk + 64 + 31) >> 5;   // key blocks needed by the last wave

  issue_tile_dma(kbf, vT, ldsk[0], ldsv[0], b, 0, tid);   // prologue: tile 0

  for (int t = 0; t < njb; ++t) {
    const int jb = t * 32;

    if (t + 1 < njb) {                     // stream next tile behind compute
      issue_tile_dma(kbf, vT, ldsk[(t + 1) & 1], ldsv[(t + 1) & 1],
                     b, jb + 32, tid);
      asm volatile("s_wait_asynccnt 0x4" ::: "memory");   // tile t resident
    } else {
      asm volatile("s_wait_asynccnt 0x0" ::: "memory");
    }
    __syncthreads();                       // visible to all 4 waves

    const __bf16* lk = ldsk[t & 1];
    const __bf16* lv = ldsv[t & 1];

    if (jb <= ibase + 15) {                // wave-uniform causal guard
      // S = Q K^T for key columns [jb, jb+32), B-fragments from LDS
      v8f s0 = {}, s1 = {};
#pragma unroll
      for (int kk = 0; kk < 2; ++kk) {
        const v16bf kf0 = *(const v16bf*)(&lk[(m)      * 64 + kk * 32 + half * 16]);
        const v16bf kf1 = *(const v16bf*)(&lk[(16 + m) * 64 + kk * 32 + half * 16]);
        s0 = wmma_bf16(qa[kk].v, kf0, s0);
        s1 = wmma_bf16(qa[kk].v, kf1, s1);
      }

      const bool edge = (jb + 31 > ibase); // tile touches the diagonal
#pragma unroll
      for (int r = 0; r < 8; ++r) {
        const int i = ibase + r + half * 8;   // this lane's query row (D layout)
        float a0 = s0[r] * kScale;
        float a1 = s1[r] * kScale;
        if (edge) {
          if (jb + m > i)      a0 = -__builtin_inff();
          if (jb + 16 + m > i) a1 = -__builtin_inff();
        }
        float tmax = fmaxf(a0, a1);
#pragma unroll
        for (int off = 1; off < 16; off <<= 1)
          tmax = fmaxf(tmax, __shfl_xor(tmax, off, 16));
        const float newm  = fmaxf(mrow[r], tmax);
        const float alpha = __expf(mrow[r] - newm);
        mrow[r] = newm;
        const float p0 = __expf(a0 - newm);
        const float p1 = __expf(a1 - newm);
        float ps = p0 + p1;
#pragma unroll
        for (int off = 1; off < 16; off <<= 1)
          ps += __shfl_xor(ps, off, 16);
        lrow[r] = lrow[r] * alpha + ps;
#pragma unroll
        for (int c = 0; c < 4; ++c) acc[c][r] *= alpha;
        pshm[wave][(r + half * 8) * 32 + m]      = (__bf16)p0;
        pshm[wave][(r + half * 8) * 32 + 16 + m] = (__bf16)p1;
      }

      asm volatile("s_wait_dscnt 0" ::: "memory");  // P visible to this wave
      V16U pa;
      pa.h[0] = *(const v8bf*)(&pshm[wave][m * 32 + half * 16]);
      pa.h[1] = *(const v8bf*)(&pshm[wave][m * 32 + half * 16 + 8]);

      // out += P V  (V B-fragment: contiguous key index per h-column lane)
#pragma unroll
      for (int c = 0; c < 4; ++c) {
        const v16bf vf = *(const v16bf*)(&lv[(c * 16 + m) * 32 + half * 16]);
        acc[c] = wmma_bf16(pa.v, vf, acc[c]);
      }
    }

    // All waves' LDS reads of buffer (t&1) must finish before iteration t+1
    // issues the DMA that overwrites it.
    asm volatile("s_wait_dscnt 0" ::: "memory");
    __syncthreads();
  }

#pragma unroll
  for (int r = 0; r < 8; ++r) {
    const float inv = 1.0f / lrow[r];
#pragma unroll
    for (int c = 0; c < 4; ++c)
      out[(size_t)(row0 + r + half * 8) * kH + c * 16 + m] = acc[c][r] * inv;
  }
}

// ---------------------------------------------------------------------------
extern "C" void kernel_launch(void* const* d_in, const int* in_sizes, int n_in,
                              void* d_out, int out_size, void* d_ws, size_t ws_size,
                              hipStream_t stream) {
  // setup_inputs order: x, Wk, Wq, Wv
  const float* x  = (const float*)d_in[0];
  const float* Wk = (const float*)d_in[1];
  const float* Wq = (const float*)d_in[2];
  const float* Wv = (const float*)d_in[3];
  float* out = (float*)d_out;

  // Workspace layout (bf16): wT[3][64][1024] | Q[16384][64] | K[16384][64] | Vt[64][16384]
  // Total = (3*65536 + 3*1048576) * 2 bytes ≈ 6.4 MB
  __bf16* wT  = (__bf16*)d_ws;
  __bf16* qbf = wT + 3 * kC * kH;
  __bf16* kbf = qbf + (size_t)kRows * kH;
  __bf16* vT  = kbf + (size_t)kRows * kH;

  wtrans_kernel<<<(3 * kC * kH) / 256, 256, 0, stream>>>(Wq, Wk, Wv, wT);
  proj_kernel<<<dim3(kRows / 16, 3), 32, 0, stream>>>(x, wT, qbf, kbf, vT);
  attn_kernel<<<kRows / 64, 128, 0, stream>>>(qbf, kbf, vT, out);
}
